// MultiHeadAttention_55422257988093
// MI455X (gfx1250) — matrix operand order repaired and checked by reading, not compiled, not run on hardware
//
#include <hip/hip_runtime.h>

// ---------------------------------------------------------------------------
// MultiHeadAttention for MI455X (gfx1250, wave32, WMMA bf16 path)
// B=4, S=2048, H=16, D=64, DM=1024
// ---------------------------------------------------------------------------

typedef __bf16 bf16_t;
typedef __attribute__((ext_vector_type(16))) __bf16 v16bf;
typedef __attribute__((ext_vector_type(8)))  __bf16 v8bf;
typedef __attribute__((ext_vector_type(8)))  float  v8f;
typedef __attribute__((ext_vector_type(4)))  float  v4f;
typedef __attribute__((ext_vector_type(4)))  int    v4i;

static constexpr int Bsz = 4;
static constexpr int Sq  = 2048;
static constexpr int Hh  = 16;
static constexpr int Dd  = 64;
static constexpr int DMc = 1024;
static constexpr int Mtot = Bsz * Sq;          // 8192 rows
static constexpr int KSTEPS = DMc / 32;        // 32 K-steps in the GEMMs

// Async global->LDS staging path (CDNA5 GLOBAL_LOAD_ASYNC_TO_LDS_B128,
// ASYNCcnt-tracked). Guarded: falls back to direct global loads if this
// toolchain doesn't expose the builtin.
#if defined(__gfx1250__) && __has_builtin(__builtin_amdgcn_global_load_async_to_lds_b128)
#define USE_ASYNC_LDS 1
#else
#define USE_ASYNC_LDS 0
#endif

// ----- WMMA wrapper: D = A(16x32 bf16) * B(32x16 bf16) + C(16x16 f32) -------
__device__ __forceinline__ v8f wmma_bf16(v16bf a, v16bf b, v8f c) {
  return __builtin_amdgcn_wmma_f32_16x16x32_bf16(
      /*neg_a=*/false, a, /*neg_b=*/false, b,
      /*c_mod=*/(short)0, c, /*reuse_a=*/false, /*reuse_b=*/false);
}

// ----- A-matrix fragment (16x32, bf16 source, row-major, stride lda) --------
// ISA layout: lanes 0-15 row M=lane, elems 0..7 = K 0..7, elems 8..15 = K 16..23
//             lanes 16-31 row M=lane-16, elems 0..7 = K 8..15, 8..15 = K 24..31
__device__ __forceinline__ v16bf frag_A_bf16(const bf16_t* a, int lda, int lane) {
  const int m = lane & 15, hi = lane >> 4;
  const bf16_t* p = a + (size_t)m * lda + hi * 8;
  v8bf c0 = *(const v8bf*)(p);
  v8bf c1 = *(const v8bf*)(p + 16);
  v16bf r;
#pragma unroll
  for (int i = 0; i < 8; ++i) { r[i] = c0[i]; r[i + 8] = c1[i]; }
  return r;
}

// Same A fragment, but source is fp32 (converted in-register to bf16)
__device__ __forceinline__ v16bf frag_A_f32(const float* a, int lda, int lane) {
  const int m = lane & 15, hi = lane >> 4;
  const float* p = a + (size_t)m * lda + hi * 8;
  v4f c0 = *(const v4f*)(p);
  v4f c1 = *(const v4f*)(p + 4);
  v4f c2 = *(const v4f*)(p + 16);
  v4f c3 = *(const v4f*)(p + 20);
  v16bf r;
#pragma unroll
  for (int i = 0; i < 4; ++i) {
    r[i]      = (__bf16)c0[i];
    r[i + 4]  = (__bf16)c1[i];
    r[i + 8]  = (__bf16)c2[i];
    r[i + 12] = (__bf16)c3[i];
  }
  return r;
}

// ----- B-matrix fragment (32x16): B[k][n] read from src where "row n" has ---
// contiguous k (i.e. src + n*ldb + k). Lanes 0-15: K 0..7,16..23, lanes 16-31: K 8..15,24..31,
// N = lane & 15.  Two 16-byte loads per lane (16B alignment always guaranteed;
// avoids over-aligned 32B vector derefs on 80B-pitch LDS tiles).
__device__ __forceinline__ v16bf frag_B_rowK(const bf16_t* b, int ldb, int lane) {
  const int n = lane & 15, hi = lane >> 4;
  const bf16_t* p = b + (size_t)n * ldb + hi * 8;
  v8bf c0 = *(const v8bf*)(p);
  v8bf c1 = *(const v8bf*)(p + 16);
  v16bf r;
#pragma unroll
  for (int i = 0; i < 8; ++i) { r[i] = c0[i]; r[i + 8] = c1[i]; }
  return r;
}

#if USE_ASYNC_LDS
// Builtin signature (from hipcc diagnostic): param0 is int4 in addrspace(1).
__device__ __forceinline__ void async_cp16(const bf16_t* gsrc, bf16_t* lds_dst) {
  __builtin_amdgcn_global_load_async_to_lds_b128(
      (__attribute__((address_space(1))) v4i*)gsrc,
      (__attribute__((address_space(3))) v4i*)lds_dst,
      /*offset=*/0, /*cpol=*/0);
}
#endif

// ---------------------------------------------------------------------------
// fp32 -> bf16 conversion (weights only; 4 x 1M elements)
// ---------------------------------------------------------------------------
__global__ void cvt_f32_bf16(const float* __restrict__ src, bf16_t* __restrict__ dst, int n) {
  int i = blockIdx.x * blockDim.x + threadIdx.x;
  if (i < n) dst[i] = (__bf16)src[i];
}

// ---------------------------------------------------------------------------
// GEMM: out[m][n] = (sum_k A[m][k] * W[n][k] + bias[n]) * oscale
// M=8192, N=DM=1024, K=DM=1024.
// Block: 256 threads = 8 waves; block tile 128(M) x 128(N); wave tile 32x64.
// W tile (128x32 bf16) staged in LDS via async copy, double buffered.
// MODE 0: store bf16 to [B,H,S,D]   (Q, K projections; Q folds 1/sqrt(D))
// MODE 1: store bf16 to [B,H,D,S]   (V projection, pre-transposed for PV wmma)
// MODE 2: A is bf16, store fp32 to [M,DM]  (output projection)
// ---------------------------------------------------------------------------
template <int MODE>
__global__ __launch_bounds__(256) void gemm_kernel(const void* __restrict__ Av,
                                                   const bf16_t* __restrict__ W,
                                                   const float* __restrict__ bias,
                                                   void* __restrict__ outv,
                                                   float oscale) {
  const int lane = threadIdx.x & 31;
  const int w    = threadIdx.x >> 5;
  const int wm   = w >> 1;                   // 0..3  (M direction)
  const int wn   = w & 1;                    // 0..1  (N direction)
  const int m0   = blockIdx.y * 128 + wm * 32;
  const int nblk = blockIdx.x * 128;
  const int n0   = nblk + wn * 64;

  v8f acc[2][4];
#pragma unroll
  for (int i = 0; i < 2; ++i)
#pragma unroll
    for (int j = 0; j < 4; ++j) acc[i][j] = (v8f){0.f, 0.f, 0.f, 0.f, 0.f, 0.f, 0.f, 0.f};

#if USE_ASYNC_LDS
  // 2 buffers x 128 rows(N) x 32 K bf16, row pitch 40 elems (80B: 16B-aligned
  // rows, 20-bank stagger => conflict-free b128 reads across 16 lanes).
  __shared__ bf16_t wtile[2][128 * 40];
  const int tid = threadIdx.x;
  auto stage = [&](int buf, int k0) {
    // 512 chunks of 16B (8 bf16): 4 chunks per N-row, 2 per thread.
#pragma unroll
    for (int i = 0; i < 2; ++i) {
      const int c  = tid + i * 256;
      const int n  = c >> 2;
      const int ko = (c & 3) * 8;
      async_cp16(W + (size_t)(nblk + n) * DMc + k0 + ko, &wtile[buf][n * 40 + ko]);
    }
  };
  stage(0, 0);
  for (int ks = 0; ks < KSTEPS; ++ks) {
    const int k0 = ks * 32;
    if (ks + 1 < KSTEPS) {
      stage((ks + 1) & 1, k0 + 32);
      asm volatile("s_wait_asynccnt 0x2" ::: "memory");   // current stage done
    } else {
      asm volatile("s_wait_asynccnt 0x0" ::: "memory");
    }
    __syncthreads();
    const bf16_t* wt = &wtile[ks & 1][0];
    v16bf af[2];
#pragma unroll
    for (int i = 0; i < 2; ++i) {
      if constexpr (MODE == 2)
        af[i] = frag_A_bf16((const bf16_t*)Av + (size_t)(m0 + i * 16) * DMc + k0, DMc, lane);
      else
        af[i] = frag_A_f32((const float*)Av + (size_t)(m0 + i * 16) * DMc + k0, DMc, lane);
    }
#pragma unroll
    for (int j = 0; j < 4; ++j) {
      v16bf bfr = frag_B_rowK(wt + (size_t)(wn * 64 + j * 16) * 40, 40, lane);
#pragma unroll
      for (int i = 0; i < 2; ++i) acc[i][j] = wmma_bf16(af[i], bfr, acc[i][j]);
    }
    __syncthreads();   // all waves done with this buffer before it is refilled
  }
#else
  for (int k0 = 0; k0 < DMc; k0 += 32) {
    v16bf af[2];
#pragma unroll
    for (int i = 0; i < 2; ++i) {
      if constexpr (MODE == 2)
        af[i] = frag_A_bf16((const bf16_t*)Av + (size_t)(m0 + i * 16) * DMc + k0, DMc, lane);
      else
        af[i] = frag_A_f32((const float*)Av + (size_t)(m0 + i * 16) * DMc + k0, DMc, lane);
    }
#pragma unroll
    for (int j = 0; j < 4; ++j) {
      v16bf bfr = frag_B_rowK(W + (size_t)(n0 + j * 16) * DMc + k0, DMc, lane);
#pragma unroll
      for (int i = 0; i < 2; ++i) acc[i][j] = wmma_bf16(af[i], bfr, acc[i][j]);
    }
  }
#endif

  // Epilogue: C layout -> row m = m0+i*16+hi*8+r, col n = n0+j*16+(lane&15)
  const int hi = lane >> 4, nl = lane & 15;
#pragma unroll
  for (int i = 0; i < 2; ++i) {
#pragma unroll
    for (int j = 0; j < 4; ++j) {
      const int n  = n0 + j * 16 + nl;
      const float bv = bias[n];
#pragma unroll
      for (int r = 0; r < 8; ++r) {
        const int m    = m0 + i * 16 + hi * 8 + r;
        const float val = (acc[i][j][r] + bv) * oscale;
        if constexpr (MODE == 2) {
          ((float*)outv)[(size_t)m * DMc + n] = val;
        } else {
          const int b = m >> 11, s = m & (Sq - 1);
          const int h = n >> 6,  d = n & 63;
          size_t dst;
          if constexpr (MODE == 0)
            dst = (((size_t)(b * Hh + h)) * Sq + s) * Dd + d;     // [B,H,S,D]
          else
            dst = (((size_t)(b * Hh + h)) * Dd + d) * Sq + s;     // [B,H,D,S]
          ((bf16_t*)outv)[dst] = (__bf16)val;
        }
      }
    }
  }
}

// ---------------------------------------------------------------------------
// Flash attention (causal), one (b,h, 64-row q block) per workgroup.
// 4 waves; each wave owns a 16-row q tile; online softmax over 32-wide kv tiles.
// 1/sqrt(D) is pre-folded into the Q projection.
// Qb, Kb: [B,H,S,D] bf16.  Vt: [B,H,D,S] bf16.  Ob: [B,S,DM] bf16.
// ---------------------------------------------------------------------------
__global__ __launch_bounds__(128) void attn_kernel(const bf16_t* __restrict__ Qb,
                                                   const bf16_t* __restrict__ Kb,
                                                   const bf16_t* __restrict__ Vt,
                                                   bf16_t* __restrict__ Ob,
                                                   const int* __restrict__ maskp) {
  __shared__ bf16_t plds[4][16 * 40];     // per-wave 16x32 P tile, pitch 40

  const int lane = threadIdx.x & 31;
  const int w    = threadIdx.x >> 5;
  const int bh   = blockIdx.y;            // 0..63
  const int q0   = blockIdx.x * 64 + w * 16;
  const int hi   = lane >> 4, nl = lane & 15;

  const bf16_t* Q = Qb + (size_t)bh * Sq * Dd;
  const bf16_t* K = Kb + (size_t)bh * Sq * Dd;
  const bf16_t* V = Vt + (size_t)bh * Dd * Sq;
  const bool causal = (*maskp) != 0;

  // Q fragments: A(16x32) over d in {0..31, 32..63}
  v16bf qa0 = frag_A_bf16(Q + (size_t)q0 * Dd + 0,  Dd, lane);
  v16bf qa1 = frag_A_bf16(Q + (size_t)q0 * Dd + 32, Dd, lane);

  v8f o[4];
#pragma unroll
  for (int t = 0; t < 4; ++t) o[t] = (v8f){0.f, 0.f, 0.f, 0.f, 0.f, 0.f, 0.f, 0.f};
  float mrun[8], lrun[8];
#pragma unroll
  for (int r = 0; r < 8; ++r) { mrun[r] = -__builtin_inff(); lrun[r] = 0.f; }

  bf16_t* pl = plds[w];
  const int jmax = causal ? (q0 + 15) / 32 : (Sq / 32 - 1);

  for (int j = 0; j <= jmax; ++j) {
    const int kv0 = j * 32;

    // Prefetch next kv tile of K (32 rows x 128B) and V (64 rows x 64B)
    if (j < jmax) {
      __builtin_prefetch(K + (size_t)(kv0 + 32 + lane) * Dd, 0, 1);
      __builtin_prefetch(V + (size_t)(2 * lane + 0) * Sq + kv0 + 32, 0, 1);
      __builtin_prefetch(V + (size_t)(2 * lane + 1) * Sq + kv0 + 32, 0, 1);
    }

    // S tile = Q(16x64) * K^T(64x32), two 16-col halves, contraction d in 2 steps
    v8f s0 = (v8f){0.f, 0.f, 0.f, 0.f, 0.f, 0.f, 0.f, 0.f};
    v8f s1 = (v8f){0.f, 0.f, 0.f, 0.f, 0.f, 0.f, 0.f, 0.f};
#pragma unroll
    for (int dd = 0; dd < 2; ++dd) {
      v16bf qa = dd ? qa1 : qa0;
      v16bf kb0 = frag_B_rowK(K + (size_t)(kv0 + 0)  * Dd + dd * 32, Dd, lane);
      v16bf kb1 = frag_B_rowK(K + (size_t)(kv0 + 16) * Dd + dd * 32, Dd, lane);
      s0 = wmma_bf16(qa, kb0, s0);
      s1 = wmma_bf16(qa, kb1, s1);
    }

    const bool need_mask = causal && (kv0 + 31 > q0);
    float alpha[8];
#pragma unroll
    for (int r = 0; r < 8; ++r) {
      const int row = q0 + hi * 8 + r;
      float a = s0[r];
      float b = s1[r];
      if (need_mask) {
        if (kv0 + nl > row)      a = -1e9f;
        if (kv0 + 16 + nl > row) b = -1e9f;
      }
      // row max over the 16 lanes holding this row (xor<16 stays in half-group)
      float mx = fmaxf(a, b);
#pragma unroll
      for (int off = 8; off >= 1; off >>= 1) mx = fmaxf(mx, __shfl_xor(mx, off, 32));
      const float mnew = fmaxf(mrun[r], mx);
      alpha[r] = __expf(mrun[r] - mnew);        // 0 on first iter (exp(-inf))
      const float pa = __expf(a - mnew);
      const float pb = __expf(b - mnew);
      float sum = pa + pb;
#pragma unroll
      for (int off = 8; off >= 1; off >>= 1) sum += __shfl_xor(sum, off, 32);
      lrun[r] = lrun[r] * alpha[r] + sum;
      mrun[r] = mnew;
      // stage P into this wave's LDS region (C layout -> row-major 16x32)
      pl[(hi * 8 + r) * 40 + nl]      = (__bf16)pa;
      pl[(hi * 8 + r) * 40 + 16 + nl] = (__bf16)pb;
    }

    // rescale running output
#pragma unroll
    for (int t = 0; t < 4; ++t)
#pragma unroll
      for (int r = 0; r < 8; ++r) o[t][r] *= alpha[r];

    // wave-private LDS RAW: DS ops are in-order per wave; dscnt==0 => visible
    asm volatile("s_wait_dscnt 0x0" ::: "memory");

    // P as A fragment (16x32), then O(16x64) += P * V(32x64) in 4 N-tiles
    v16bf pfrag = frag_A_bf16(pl, 40, lane);
#pragma unroll
    for (int t = 0; t < 4; ++t) {
      v16bf vb = frag_B_rowK(V + (size_t)(t * 16) * Sq + kv0, Sq, lane);
      o[t] = wmma_bf16(pfrag, vb, o[t]);
    }
  }

  // normalize + store to attn-out [B,S,DM] bf16
  const int b = bh >> 4, h = bh & 15;
#pragma unroll
  for (int t = 0; t < 4; ++t) {
#pragma unroll
    for (int r = 0; r < 8; ++r) {
      const int s = q0 + hi * 8 + r;
      const float val = o[t][r] / lrun[r];
      Ob[((size_t)(b * Sq + s)) * DMc + h * 64 + t * 16 + nl] = (__bf16)val;
    }
  }
}

// ---------------------------------------------------------------------------
extern "C" void kernel_launch(void* const* d_in, const int* in_sizes, int n_in,
                              void* d_out, int out_size, void* d_ws, size_t ws_size,
                              hipStream_t stream) {
  const float* q  = (const float*)d_in[0];
  const float* k  = (const float*)d_in[1];
  const float* v  = (const float*)d_in[2];
  const float* Wq = (const float*)d_in[3];
  const float* bq = (const float*)d_in[4];
  const float* Wk = (const float*)d_in[5];
  const float* bk = (const float*)d_in[6];
  const float* Wv = (const float*)d_in[7];
  const float* bv = (const float*)d_in[8];
  const float* Wo = (const float*)d_in[9];
  const float* bo = (const float*)d_in[10];
  const int* masking = (const int*)d_in[11];

  // workspace layout (72 MB total)
  bf16_t* Wqb = (bf16_t*)d_ws;
  bf16_t* Wkb = Wqb + (size_t)DMc * DMc;
  bf16_t* Wvb = Wkb + (size_t)DMc * DMc;
  bf16_t* Wob = Wvb + (size_t)DMc * DMc;
  bf16_t* Qb  = Wob + (size_t)DMc * DMc;           // [B,H,S,D]
  bf16_t* Kb  = Qb  + (size_t)Mtot * DMc;          // [B,H,S,D]
  bf16_t* Vtb = Kb  + (size_t)Mtot * DMc;          // [B,H,D,S]
  bf16_t* AOb = Vtb + (size_t)Mtot * DMc;          // [B,S,DM]

  const int nW = DMc * DMc;
  cvt_f32_bf16<<<(nW + 255) / 256, 256, 0, stream>>>(Wq, Wqb, nW);
  cvt_f32_bf16<<<(nW + 255) / 256, 256, 0, stream>>>(Wk, Wkb, nW);
  cvt_f32_bf16<<<(nW + 255) / 256, 256, 0, stream>>>(Wv, Wvb, nW);
  cvt_f32_bf16<<<(nW + 255) / 256, 256, 0, stream>>>(Wo, Wob, nW);

  dim3 gg(DMc / 128, Mtot / 128);                  // 8 x 64 blocks
  gemm_kernel<0><<<gg, 256, 0, stream>>>(q, Wqb, bq, Qb,  0.125f);  // fold 1/sqrt(D)
  gemm_kernel<0><<<gg, 256, 0, stream>>>(k, Wkb, bk, Kb,  1.0f);
  gemm_kernel<1><<<gg, 256, 0, stream>>>(v, Wvb, bv, Vtb, 1.0f);

  attn_kernel<<<dim3(Sq / 64, Bsz * Hh), 128, 0, stream>>>(Qb, Kb, Vtb, AOb, masking);

  gemm_kernel<2><<<gg, 256, 0, stream>>>(AOb, Wob, bo, d_out, 1.0f);
}
